// EagerAttention_42004780154927
// MI455X (gfx1250) — matrix operand order repaired and checked by reading, not compiled, not run on hardware
//
#include <hip/hip_runtime.h>
#include <hip/hip_bf16.h>

// ---------------------------------------------------------------------------
// EagerAttention for MI455X (gfx1250): bf16 WMMA pipeline with double-buffered
// Tensor-Data-Mover tile staging (DMA of tile i+1 overlaps WMMAs on tile i).
//   B=2 S=2048 E=1024 H=16 D=64, scale = 1/8
// ---------------------------------------------------------------------------

typedef __bf16 bf16_t;
typedef __attribute__((ext_vector_type(16))) __bf16 v16bf;
typedef __attribute__((ext_vector_type(8)))  float  v8f;
typedef __attribute__((ext_vector_type(4)))  unsigned int u32x4;
typedef __attribute__((ext_vector_type(8)))  int   i32x8;
typedef __attribute__((ext_vector_type(4)))  int   i32x4;

#define B_  2
#define S_  2048
#define E_  1024
#define H_  16
#define D_  64
#define NROWS (B_ * S_)   // 4096 tokens

#if defined(__has_builtin)
#if __has_builtin(__builtin_amdgcn_tensor_load_to_lds) && \
    __has_builtin(__builtin_amdgcn_s_wait_tensorcnt)
#define HAVE_TDM 1
#endif
#endif
#ifndef HAVE_TDM
#define HAVE_TDM 0
#endif

#if HAVE_TDM
// Tensor Data Mover: DMA a 2D tile (tile_rows x tile_cols, 2-byte elements)
// from a row-major tensor (row stride in elements) into a packed LDS tile.
// D# bit layout per CDNA5 ISA ch.10.8 (group0 128b, group1 256b).
// 6-arg builtin: (u32x4 g0, i32x8 g1, i32x4 g2, i32x4 g3, i32x8 g4, i32 cpol)
__device__ __forceinline__ void tdm_load_2d(const void* gptr, unsigned lds_off,
                                            unsigned tile_cols,
                                            unsigned tile_rows,
                                            unsigned tensor_cols,
                                            unsigned tensor_rows,
                                            unsigned row_stride_elems) {
  unsigned long long ga = (unsigned long long)(uintptr_t)gptr;
  u32x4 g0;
  g0.x = 1u;                                   // count=1, user descriptor
  g0.y = lds_off;                              // lds_addr (bytes)
  g0.z = (unsigned)ga;                         // global_addr[31:0]
  g0.w = (unsigned)((ga >> 32) & 0x01FFFFFFu)  // global_addr[56:32]
         | (2u << 30);                         // type=2 ("image")
  i32x8 g1;
  g1[0] = (int)(1u << 16);                     // data_size=1 -> 2 bytes
  g1[1] = (int)((tensor_cols & 0xFFFFu) << 16);          // tensor_dim0[15:0]
  g1[2] = (int)((tensor_cols >> 16) |                    // tensor_dim0[31:16]
                ((tensor_rows & 0xFFFFu) << 16));        // tensor_dim1[15:0]
  g1[3] = (int)(((tensor_rows >> 16) & 0xFFFFu) |        // tensor_dim1[31:16]
                (tile_cols << 16));                      // tile_dim0
  g1[4] = (int)(tile_rows & 0xFFFFu);                    // tile_dim1 (dim2=0)
  g1[5] = (int)row_stride_elems;                         // dim0_stride[31:0]
  g1[6] = 0;                                             // stride hi, dim1_str
  g1[7] = 0;
  i32x4 z4 = {0, 0, 0, 0};
  i32x8 z8 = {0, 0, 0, 0, 0, 0, 0, 0};
  __builtin_amdgcn_tensor_load_to_lds(g0, g1, z4, z4, z8, 0);
}
#endif

__device__ __forceinline__ unsigned lds_offset_of(const void* p) {
  // Flat LDS aperture places the workgroup-relative offset in addr[31:0].
  return (unsigned)(uintptr_t)p;
}

// ---------------------------------------------------------------------------
// fp32 -> bf16 elementwise convert
// ---------------------------------------------------------------------------
__global__ void cvt_f32_bf16(const float* __restrict__ in,
                             bf16_t* __restrict__ out, int n) {
  int i = blockIdx.x * blockDim.x + threadIdx.x;
  int stride = gridDim.x * blockDim.x;
  for (; i < n; i += stride) out[i] = (bf16_t)in[i];
}

// ---------------------------------------------------------------------------
// NT GEMM:  C[m,n] = sum_k A[m,k] * Bw[n,k]     (A, Bw row-major bf16)
// Tile 128x128, K-step 32, 8 wave32 waves (256 thr), 2x4 WMMA frags per wave.
// TDM ping-pong: tile i+1 DMA in flight while tile i runs on the WMMA units;
// tensor ops retire in order per wave, so wait_tensorcnt(1) covers tile i.
// ---------------------------------------------------------------------------
#define TM 128
#define TN 128
#define TK 32

template <bool OUT_BF16>
__global__ __launch_bounds__(256) void gemm_nt(const bf16_t* __restrict__ A,
                                               const bf16_t* __restrict__ Bw,
                                               void* __restrict__ Cp,
                                               int M, int N, int K) {
  __shared__ __align__(32) bf16_t Asm[2][TM * TK];
  __shared__ __align__(32) bf16_t Bsm[2][TN * TK];

  const int t    = threadIdx.x;
  const int lane = t & 31;
  const int wave = t >> 5;      // 0..7
  const int wm   = wave & 3;    // 4 row-groups  -> 32 rows each
  const int wn   = wave >> 2;   // 2 col-groups  -> 64 cols each
  const int bm   = blockIdx.x * TM;
  const int bn   = blockIdx.y * TN;
  const int lo   = lane & 15;
  const int hi   = lane >> 4;

  const v8f vzero = {0.f, 0.f, 0.f, 0.f, 0.f, 0.f, 0.f, 0.f};
  v8f acc[2][4];
#pragma unroll
  for (int i = 0; i < 2; ++i)
#pragma unroll
    for (int j = 0; j < 4; ++j) acc[i][j] = vzero;

#if HAVE_TDM
  const unsigned ldsA[2] = {lds_offset_of(&Asm[0][0]),
                            lds_offset_of(&Asm[1][0])};
  const unsigned ldsB[2] = {lds_offset_of(&Bsm[0][0]),
                            lds_offset_of(&Bsm[1][0])};
  // Prologue: DMA tile 0 into buffer 0.
  if (wave == 0)
    tdm_load_2d(&A[(size_t)bm * K], ldsA[0], TK, TM, K, M, K);
  else if (wave == 1)
    tdm_load_2d(&Bw[(size_t)bn * K], ldsB[0], TK, TN, K, N, K);
#endif

  int it = 0;
  for (int k0 = 0; k0 < K; k0 += TK, ++it) {
    const int cur = it & 1;
#if HAVE_TDM
    const int nxt = cur ^ 1;
    if (k0 + TK < K) {
      // Issue DMA for tile i+1; previous-iteration barrier guarantees the
      // target buffer is no longer being read by any wave.
      if (wave == 0)
        tdm_load_2d(&A[(size_t)bm * K + k0 + TK], ldsA[nxt], TK, TM, K, M, K);
      else if (wave == 1)
        tdm_load_2d(&Bw[(size_t)bn * K + k0 + TK], ldsB[nxt], TK, TN, K, N, K);
      __builtin_amdgcn_s_wait_tensorcnt(1);  // oldest DMA (tile i) complete
      if (k0 + 2 * TK < K) {  // warm L2 two tiles ahead (global_prefetch_b8)
        __builtin_prefetch(&A[(size_t)bm * K + k0 + 2 * TK], 0, 1);
        __builtin_prefetch(&Bw[(size_t)bn * K + k0 + 2 * TK], 0, 1);
      }
    } else {
      __builtin_amdgcn_s_wait_tensorcnt(0);  // drain on last tile
    }
#else
    // Fallback: cooperative staging, 8 bf16 (16B) per thread per pass.
#pragma unroll
    for (int p = 0; p < 2; ++p) {
      int e   = (p * 256 + t) * 8;
      int row = e >> 5;
      int col = e & 31;
      *(uint4*)&Asm[cur][row * TK + col] =
          *(const uint4*)&A[(size_t)(bm + row) * K + k0 + col];
      *(uint4*)&Bsm[cur][row * TK + col] =
          *(const uint4*)&Bw[(size_t)(bn + row) * K + k0 + col];
    }
#endif
    __syncthreads();

    v16bf af[2];
#pragma unroll
    for (int i = 0; i < 2; ++i)
      af[i] =
          *(const v16bf*)&Asm[cur][(wm * 32 + i * 16 + lo) * TK + hi * 16];
#pragma unroll
    for (int j = 0; j < 4; ++j) {
      v16bf bfr =
          *(const v16bf*)&Bsm[cur][(wn * 64 + j * 16 + lo) * TK + hi * 16];
#pragma unroll
      for (int i = 0; i < 2; ++i)
        acc[i][j] = __builtin_amdgcn_wmma_f32_16x16x32_bf16(
            false, af[i], false, bfr, (short)0, acc[i][j], false, false);
    }
    __syncthreads();
  }

  // C layout: VGPR r, lane -> (Mrow = r + 8*hi, Ncol = lo)
#pragma unroll
  for (int i = 0; i < 2; ++i)
#pragma unroll
    for (int j = 0; j < 4; ++j)
#pragma unroll
      for (int r = 0; r < 8; ++r) {
        int gm = bm + wm * 32 + i * 16 + r + (hi << 3);
        int gn = bn + wn * 64 + j * 16 + lo;
        if (OUT_BF16)
          ((bf16_t*)Cp)[(size_t)gm * N + gn] = (bf16_t)acc[i][j][r];
        else
          ((float*)Cp)[(size_t)gm * N + gn] = acc[i][j][r];
      }
}

// ---------------------------------------------------------------------------
// Flash attention: per (b,h), BR=64 query rows/block (4 waves x 16 rows),
// streaming BC=64 key/value rows through LDS with online softmax.
// K/V tiles double-buffered via TDM; layout [B,S,H,D] == [row, h*64+d].
// ---------------------------------------------------------------------------
#define BR 64
#define BC 64

__global__ __launch_bounds__(128) void flash_attn(const bf16_t* __restrict__ Q,
                                                  const bf16_t* __restrict__ Kb,
                                                  const bf16_t* __restrict__ V,
                                                  bf16_t* __restrict__ O,
                                                  float scale) {
  __shared__ __align__(32) bf16_t Ksm[2][BC * D_];
  __shared__ __align__(32) bf16_t Vsm[2][BC * D_];
  __shared__ __align__(32) bf16_t Psm[4][16 * BC];

  const int t    = threadIdx.x;
  const int lane = t & 31;
  const int wave = t >> 5;                 // 0..3
  const int lo   = lane & 15;
  const int hi   = lane >> 4;
  const int h    = blockIdx.y;
  const int b    = blockIdx.z;

  const size_t rowS    = E_;
  const size_t headOff = (size_t)h * D_;
  const size_t baseRow = (size_t)b * S_;
  const int    qr0     = blockIdx.x * BR + wave * 16;

  // Q fragments pinned in registers for the whole key loop (16x64 per wave).
  v16bf qa[2];
#pragma unroll
  for (int kk = 0; kk < 2; ++kk)
    qa[kk] = *(const v16bf*)&Q[(baseRow + qr0 + lo) * rowS + headOff +
                               kk * 32 + hi * 16];

  const v8f vzero = {0.f, 0.f, 0.f, 0.f, 0.f, 0.f, 0.f, 0.f};
  float mrow[8], lrow[8];
  v8f   o[4];
#pragma unroll
  for (int r = 0; r < 8; ++r) { mrow[r] = -3.0e38f; lrow[r] = 0.f; }
#pragma unroll
  for (int j = 0; j < 4; ++j) o[j] = vzero;

#if HAVE_TDM
  const unsigned ldsK[2] = {lds_offset_of(&Ksm[0][0]),
                            lds_offset_of(&Ksm[1][0])};
  const unsigned ldsV[2] = {lds_offset_of(&Vsm[0][0]),
                            lds_offset_of(&Vsm[1][0])};
  // Prologue: DMA key-block 0 into buffer 0.
  if (wave == 0)
    tdm_load_2d(&Kb[baseRow * rowS + headOff], ldsK[0], D_, BC, E_, NROWS, E_);
  else if (wave == 1)
    tdm_load_2d(&V[baseRow * rowS + headOff], ldsV[0], D_, BC, E_, NROWS, E_);
#endif

  int it = 0;
  for (int kb = 0; kb < S_; kb += BC, ++it) {
    const int cur = it & 1;
#if HAVE_TDM
    const int nxt = cur ^ 1;
    if (kb + BC < S_) {
      if (wave == 0)
        tdm_load_2d(&Kb[(baseRow + kb + BC) * rowS + headOff], ldsK[nxt], D_,
                    BC, E_, NROWS, E_);
      else if (wave == 1)
        tdm_load_2d(&V[(baseRow + kb + BC) * rowS + headOff], ldsV[nxt], D_,
                    BC, E_, NROWS, E_);
      __builtin_amdgcn_s_wait_tensorcnt(1);  // key-block kb complete
    } else {
      __builtin_amdgcn_s_wait_tensorcnt(0);
    }
#else
    {  // Fallback staging: 64 rows x 64 halves; 2 threads per row, 64B each.
      int row = t >> 1;
      int seg = (t & 1) * 32;
      const bf16_t* ksrc = &Kb[(baseRow + kb + row) * rowS + headOff + seg];
      const bf16_t* vsrc = &V[(baseRow + kb + row) * rowS + headOff + seg];
#pragma unroll
      for (int q4 = 0; q4 < 4; ++q4) {
        *(uint4*)&Ksm[cur][row * D_ + seg + q4 * 8] =
            *(const uint4*)&ksrc[q4 * 8];
        *(uint4*)&Vsm[cur][row * D_ + seg + q4 * 8] =
            *(const uint4*)&vsrc[q4 * 8];
      }
    }
#endif
    __syncthreads();

    // S = Q K^T   (16 x 64 per wave, fp32 accum)
    v8f sacc[4];
#pragma unroll
    for (int j = 0; j < 4; ++j) sacc[j] = vzero;
#pragma unroll
    for (int kk = 0; kk < 2; ++kk)
#pragma unroll
      for (int j = 0; j < 4; ++j) {
        v16bf kf =
            *(const v16bf*)&Ksm[cur][(j * 16 + lo) * D_ + kk * 32 + hi * 16];
        sacc[j] = __builtin_amdgcn_wmma_f32_16x16x32_bf16(
            false, qa[kk], false, kf, (short)0, sacc[j], false, false);
      }

    // Online softmax: per-register row r lives in a 16-lane half; xor masks
    // 1/2/4/8 reduce within that half (wave32).
#pragma unroll
    for (int r = 0; r < 8; ++r) {
      float s0 = sacc[0][r] * scale, s1 = sacc[1][r] * scale;
      float s2 = sacc[2][r] * scale, s3 = sacc[3][r] * scale;
      float mloc = fmaxf(fmaxf(s0, s1), fmaxf(s2, s3));
#pragma unroll
      for (int msk = 1; msk <= 8; msk <<= 1)
        mloc = fmaxf(mloc, __shfl_xor(mloc, msk, 32));
      float mnew = fmaxf(mrow[r], mloc);
      float corr = __expf(mrow[r] - mnew);
      mrow[r] = mnew;
      float p0 = __expf(s0 - mnew), p1 = __expf(s1 - mnew);
      float p2 = __expf(s2 - mnew), p3 = __expf(s3 - mnew);
      float ls = p0 + p1 + p2 + p3;
#pragma unroll
      for (int msk = 1; msk <= 8; msk <<= 1) ls += __shfl_xor(ls, msk, 32);
      lrow[r] = lrow[r] * corr + ls;
#pragma unroll
      for (int j = 0; j < 4; ++j) o[j][r] *= corr;
      int m = r + (hi << 3);
      Psm[wave][m * BC +  0 + lo] = (bf16_t)p0;
      Psm[wave][m * BC + 16 + lo] = (bf16_t)p1;
      Psm[wave][m * BC + 32 + lo] = (bf16_t)p2;
      Psm[wave][m * BC + 48 + lo] = (bf16_t)p3;
    }
    // per-wave LDS RAW: drain DS counter (CDNA5 split wait)
    asm volatile("s_wait_dscnt 0" ::: "memory");

    // O += P V   (P: 16xBC, V: BCx64)
#pragma unroll
    for (int kk = 0; kk < 2; ++kk) {
      v16bf pf = *(const v16bf*)&Psm[wave][lo * BC + kk * 32 + hi * 16];
#pragma unroll
      for (int j = 0; j < 4; ++j) {
        v16bf vf;
#pragma unroll
        for (int s = 0; s < 16; ++s)
          vf[s] = Vsm[cur][(kk * 32 + hi * 16 + s) * D_ + j * 16 + lo];
        o[j] = __builtin_amdgcn_wmma_f32_16x16x32_bf16(
            false, pf, false, vf, (short)0, o[j], false, false);
      }
    }
    __syncthreads();
  }

  // Normalize and write out (bf16, [B,S,H,D] layout)
#pragma unroll
  for (int r = 0; r < 8; ++r) {
    float inv = 1.0f / lrow[r];
    int   m   = r + (hi << 3);
#pragma unroll
    for (int j = 0; j < 4; ++j)
      O[(baseRow + qr0 + m) * rowS + headOff + j * 16 + lo] =
          (bf16_t)(o[j][r] * inv);
  }
}

// ---------------------------------------------------------------------------
// Launch pipeline
// ---------------------------------------------------------------------------
extern "C" void kernel_launch(void* const* d_in, const int* in_sizes, int n_in,
                              void* d_out, int out_size, void* d_ws,
                              size_t ws_size, hipStream_t stream) {
  const float* x  = (const float*)d_in[0];
  const float* wq = (const float*)d_in[1];
  const float* wk = (const float*)d_in[2];
  const float* wv = (const float*)d_in[3];
  const float* wo = (const float*)d_in[4];

  char* ws = (char*)d_ws;
  bf16_t* xb  = (bf16_t*)ws;  ws += (size_t)NROWS * E_ * sizeof(bf16_t);
  bf16_t* wqb = (bf16_t*)ws;  ws += (size_t)E_ * E_ * sizeof(bf16_t);
  bf16_t* wkb = (bf16_t*)ws;  ws += (size_t)E_ * E_ * sizeof(bf16_t);
  bf16_t* wvb = (bf16_t*)ws;  ws += (size_t)E_ * E_ * sizeof(bf16_t);
  bf16_t* wob = (bf16_t*)ws;  ws += (size_t)E_ * E_ * sizeof(bf16_t);
  bf16_t* Qb  = (bf16_t*)ws;  ws += (size_t)NROWS * E_ * sizeof(bf16_t);
  bf16_t* Kb  = (bf16_t*)ws;  ws += (size_t)NROWS * E_ * sizeof(bf16_t);
  bf16_t* Vb  = (bf16_t*)ws;  ws += (size_t)NROWS * E_ * sizeof(bf16_t);
  bf16_t* Ob  = (bf16_t*)ws;  ws += (size_t)NROWS * E_ * sizeof(bf16_t);

  cvt_f32_bf16<<<1024, 256, 0, stream>>>(x, xb, NROWS * E_);
  cvt_f32_bf16<<<512, 256, 0, stream>>>(wq, wqb, E_ * E_);
  cvt_f32_bf16<<<512, 256, 0, stream>>>(wk, wkb, E_ * E_);
  cvt_f32_bf16<<<512, 256, 0, stream>>>(wv, wvb, E_ * E_);
  cvt_f32_bf16<<<512, 256, 0, stream>>>(wo, wob, E_ * E_);

  dim3 gg(NROWS / TM, E_ / TN);  // 32 x 8 blocks
  gemm_nt<true><<<gg, 256, 0, stream>>>(xb, wqb, Qb, NROWS, E_, E_);
  gemm_nt<true><<<gg, 256, 0, stream>>>(xb, wkb, Kb, NROWS, E_, E_);
  gemm_nt<true><<<gg, 256, 0, stream>>>(xb, wvb, Vb, NROWS, E_, E_);

  flash_attn<<<dim3(S_ / BR, H_, B_), 128, 0, stream>>>(Qb, Kb, Vb, Ob,
                                                        0.125f);

  gemm_nt<false><<<gg, 256, 0, stream>>>(Ob, wob, d_out, NROWS, E_, E_);
}
